// LSTM_decoder_21534966022554
// MI455X (gfx1250) — compile-verified, matrix operand order checked by reading, not run on hardware
//
#include <hip/hip_runtime.h>
#include <hip/hip_bf16.h>
#include <stdint.h>

// ---------------------------------------------------------------------------
// LSTM decoder for MI455X (gfx1250, wave32, WMMA).
// B=1024, T_ENC=50, IN=99, H=2112, SEQ_OUT=25, 4H=8448.
//
// - Weights converted once to bf16 pre-swizzled WMMA fragments (~110 MB:
//   L2-resident across all 25 sequential steps; HBM traffic ~one pass).
// - Gate GEMMs: 128/256-row register-blocked v_wmma_f32_16x16x32_bf16 with
//   B fragments staged per-workgroup via global_load_async_to_lds_b128
//   (ASYNCcnt double-buffering) and read back with ds_load_b128.
// - LSTM cell fused into the GEMM epilogue (all 4 gates land in-lane).
// ---------------------------------------------------------------------------

typedef __attribute__((ext_vector_type(16))) __bf16 v16bf;
typedef __attribute__((ext_vector_type(8)))  float  v8f;

static constexpr int BSZ   = 1024;
static constexpr int TENC  = 50;
static constexpr int IN_N  = 99;
static constexpr int HH    = 2112;
static constexpr int SEQO  = 25;
static constexpr int KT_H  = HH / 32;       // 66 k-tiles over H
static constexpr int KT_X  = 4;             // 99 padded to 128 -> 4 k-tiles
static constexpr int NT_G  = (4 * HH) / 16; // 528 n-tiles over 4H
static constexpr int NT_O  = 7;             // 99 padded to 112 -> 7 n-tiles
static constexpr int MT_N  = BSZ / 16;      // 64 m-tiles over batch

// Fragment layouts (512 bf16 per 16x32 tile, 16 elems per lane):
//   A-frag of X(MxK):  off = ((mt*KT + kt)*32 + lane)*16 + e
//   B-frag of W^T:     off = ((nt*KT + kt)*32 + lane)*16 + e
// lane<16 : local row = lane,    K-elems e -> [0..7, 16..23]
// lane>=16: local row = lane-16, K-elems e -> [8..15, 24..31]

__device__ __forceinline__ void decode_frag(int lane, int e, int& lr, int& kk) {
    int hi = (lane >> 4) & 1;
    lr = lane & 15;
    kk = (e < 8) ? (e + (hi << 3)) : (16 + (e - 8) + (hi << 3));
}

__device__ __forceinline__ size_t a_frag_off(int b, int j, int KT) {
    int mt = b >> 4, m = b & 15, kt = j >> 5, kk = j & 31;
    int lane = m + ((kk & 8) ? 16 : 0);
    int e = (kk < 16) ? (kk & 7) : (8 + (kk & 7));
    return (((size_t)mt * KT + kt) * 32 + lane) * 16 + e;
}

__device__ __forceinline__ float sigm(float x) { return 1.0f / (1.0f + __expf(-x)); }

#define WMMA_BF16(a, b, c) \
    __builtin_amdgcn_wmma_f32_16x16x32_bf16(false, (a), false, (b), (short)0, (c), false, false)

// ---------------------------------------------------------------------------
// Setup: means over encoder states -> initial h0/h1 (bf16 A-frags) and c0/c1
// ---------------------------------------------------------------------------
__global__ void setup_state_kernel(const float* __restrict__ hs,
                                   const float* __restrict__ cs,
                                   const float* __restrict__ gts,
                                   float* __restrict__ c0, float* __restrict__ c1,
                                   __bf16* __restrict__ h0f, __bf16* __restrict__ h1f) {
    int idx = blockIdx.x * blockDim.x + threadIdx.x;
    if (idx >= BSZ * HH) return;
    int b = idx / HH, j = idx % HH;
    float sh = 0.f, sc = 0.f;
    for (int t = 0; t < TENC; ++t) {
        sh += hs[((size_t)b * TENC + t) * HH + j];
        sc += cs[((size_t)b * TENC + t) * HH + j];
    }
    float h0v = sh * (1.0f / 50.0f);
    float h1v = (sh + gts[(size_t)b * HH + j]) * (1.0f / 51.0f);
    float c0v = sc * (1.0f / 50.0f);
    c0[idx] = c0v;
    c1[idx] = c0v;
    size_t off = a_frag_off(b, j, KT_H);
    h0f[off] = (__bf16)h0v;
    h1f[off] = (__bf16)h1v;
}

// ---------------------------------------------------------------------------
// Weight/activation conversion into fragment layout (coalesced writes).
// transposed=0: src is (Nsrc x Ksrc) row-major, B[k,n] = W[n,k]  (W_ih/W_hh, p)
// transposed=1: src is (Ksrc x Nsrc) row-major, B[k,n] = W[k,n]  (W_out)
// Zero-pads k >= Ksrc and n >= Nsrc.
// ---------------------------------------------------------------------------
__global__ void convert_frag_kernel(const float* __restrict__ W, __bf16* __restrict__ dst,
                                    int Nsrc, int Ksrc, int NT, int KT, int transposed) {
    long idx = (long)blockIdx.x * blockDim.x + threadIdx.x;
    long total = (long)NT * KT * 512;
    if (idx >= total) return;
    int e = idx & 15;
    int lane = (idx >> 4) & 31;
    long fi = idx >> 9;
    int kt = (int)(fi % KT);
    int nt = (int)(fi / KT);
    int lr, kk;
    decode_frag(lane, e, lr, kk);
    int n = nt * 16 + lr;
    int k = kt * 32 + kk;
    float v = 0.f;
    if (n < Nsrc && k < Ksrc)
        v = transposed ? W[(long)k * Nsrc + n] : W[(long)n * Ksrc + k];
    dst[idx] = (__bf16)v;
}

// ---------------------------------------------------------------------------
// Fused gate-GEMM + LSTM cell.
// WG = 8 wave32; tile = 256 batch x 32 hidden, all 4 gates fused.
// Each wave computes 2 M-tiles (acc[2][4][2] = 128 VGPRs) so every staged
// B fragment feeds 2 WMMAs. B fragments (8 KB/k-step, shared by all waves)
// are staged through LDS with double-buffered global_load_async_to_lds_b128.
// ---------------------------------------------------------------------------
__global__ __launch_bounds__(256) void gemm_gates_kernel(
    const __bf16* __restrict__ xfrag, int ktx, const __bf16* __restrict__ wxfrag,
    const __bf16* __restrict__ hfrag, int kth, const __bf16* __restrict__ whfrag,
    const float* __restrict__ b_ih, const float* __restrict__ b_hh,
    float* __restrict__ c_state, __bf16* __restrict__ hout_frag) {
    __shared__ __bf16 ldsB[2][8][512];   // 2 buffers x 8 frags x 1 KB

    const int w    = threadIdx.x >> 5;
    const int lane = threadIdx.x & 31;
    const int bm   = blockIdx.x;   // 0..3  (batch / 256)
    const int ht   = blockIdx.y;   // 0..65 (hidden cols / 32)
    const int mt0  = bm * 16 + w;
    const int mt1  = mt0 + 8;
    const int ktot = ktx + kth;

    // Wave w stages B fragment f==w (gate f>>1, sub-tile f&1): 32 lanes x 32 B.
    auto issueB = [&](int buf, int kg) {
        const __bf16* wp; int KT, kt;
        if (kg < ktx) { wp = wxfrag; KT = ktx; kt = kg; }
        else          { wp = whfrag; KT = kth; kt = kg - ktx; }
        int nt = (w >> 1) * (HH / 16) + ht * 2 + (w & 1);
        const __bf16* src = wp + (((size_t)nt * KT + kt) * 32 + lane) * 16;
        unsigned dst = (unsigned)(uintptr_t)&ldsB[buf][w][lane * 16];
        asm volatile("global_load_async_to_lds_b128 %0, %1, off"
                     :: "v"(dst), "v"(src) : "memory");
        asm volatile("global_load_async_to_lds_b128 %0, %1, off offset:16"
                     :: "v"(dst), "v"(src) : "memory");
    };

    v8f acc[2][4][2] = {};
    issueB(0, 0);

    for (int kg = 0; kg < ktot; ++kg) {
        const int cur = kg & 1;
        asm volatile("s_wait_asynccnt 0x0" ::: "memory"); // my stage done
        __syncthreads();                                  // everyone's stage done
        if (kg + 1 < ktot) issueB(cur ^ 1, kg + 1);       // overlap next stage

        const __bf16 *a0p, *a1p;
        if (kg < ktx) {
            a0p = xfrag + (((size_t)mt0 * ktx + kg) * 32 + lane) * 16;
            a1p = xfrag + (((size_t)mt1 * ktx + kg) * 32 + lane) * 16;
        } else {
            int kt = kg - ktx;
            a0p = hfrag + (((size_t)mt0 * kth + kt) * 32 + lane) * 16;
            a1p = hfrag + (((size_t)mt1 * kth + kt) * 32 + lane) * 16;
        }
        if (kg + 1 < ktot) __builtin_prefetch(a0p + 512, 0, 1);
        const v16bf a0 = *(const v16bf*)a0p;
        const v16bf a1 = *(const v16bf*)a1p;
#pragma unroll
        for (int g = 0; g < 4; ++g)
#pragma unroll
            for (int n2 = 0; n2 < 2; ++n2) {
                const v16bf bb = *(const v16bf*)&ldsB[cur][g * 2 + n2][lane * 16];
                acc[0][g][n2] = WMMA_BF16(a0, bb, acc[0][g][n2]);
                acc[1][g][n2] = WMMA_BF16(a1, bb, acc[1][g][n2]);
            }
    }

    // --- Epilogue: fused LSTM cell, same (b,j) holds all 4 gates in-lane ---
#pragma unroll
    for (int n2 = 0; n2 < 2; ++n2) {
        int j = ht * 32 + n2 * 16 + (lane & 15);
        float bi = b_ih[j]          + b_hh[j];
        float bf = b_ih[HH + j]     + b_hh[HH + j];
        float bg = b_ih[2 * HH + j] + b_hh[2 * HH + j];
        float bo = b_ih[3 * HH + j] + b_hh[3 * HH + j];
#pragma unroll
        for (int mi = 0; mi < 2; ++mi) {
            int mtc = (mi == 0) ? mt0 : mt1;
#pragma unroll
            for (int r = 0; r < 8; ++r) {
                int b = mtc * 16 + r + ((lane >= 16) ? 8 : 0);
                float iv = acc[mi][0][n2][r] + bi;
                float fv = acc[mi][1][n2][r] + bf;
                float gv = acc[mi][2][n2][r] + bg;
                float ov = acc[mi][3][n2][r] + bo;
                size_t ci = (size_t)b * HH + j;
                float c_new = sigm(fv) * c_state[ci] + sigm(iv) * tanhf(gv);
                float h_new = sigm(ov) * tanhf(c_new);
                c_state[ci] = c_new;
                hout_frag[a_frag_off(b, j, KT_H)] = (__bf16)h_new;
            }
        }
    }
}

// ---------------------------------------------------------------------------
// Output GEMM: pre = h1 @ W_out + b_out + prev.  M=1024, N=99(pad 112), K=2112.
// 2 M-tiles per wave; W_out fragments (462 KB) are L2-resident.
// Writes d_out[:, t, :] and the residual stream as next step's A-fragments.
// ---------------------------------------------------------------------------
__global__ __launch_bounds__(256) void gemm_out_kernel(
    const __bf16* __restrict__ hfrag, const __bf16* __restrict__ wofrag,
    const float* __restrict__ b_out,
    const float* __restrict__ prev, int prev_stride,
    float* __restrict__ out_t, __bf16* __restrict__ prefrag) {
    const int w    = threadIdx.x >> 5;
    const int lane = threadIdx.x & 31;
    const int bm   = blockIdx.x;   // 0..3
    const int mt0  = bm * 16 + w;
    const int mt1  = mt0 + 8;

    v8f acc[2][NT_O] = {};
    for (int kt = 0; kt < KT_H; ++kt) {
        const __bf16* a0p = hfrag + (((size_t)mt0 * KT_H + kt) * 32 + lane) * 16;
        const __bf16* a1p = hfrag + (((size_t)mt1 * KT_H + kt) * 32 + lane) * 16;
        if (kt + 1 < KT_H) __builtin_prefetch(a0p + 512, 0, 1);
        const v16bf a0 = *(const v16bf*)a0p;
        const v16bf a1 = *(const v16bf*)a1p;
#pragma unroll
        for (int nt = 0; nt < NT_O; ++nt) {
            const v16bf bb = *(const v16bf*)(wofrag + (((size_t)nt * KT_H + kt) * 32 + lane) * 16);
            acc[0][nt] = WMMA_BF16(a0, bb, acc[0][nt]);
            acc[1][nt] = WMMA_BF16(a1, bb, acc[1][nt]);
        }
    }
#pragma unroll
    for (int nt = 0; nt < NT_O; ++nt) {
        int n = nt * 16 + (lane & 15);
        float bo = (n < IN_N) ? b_out[n] : 0.f;
#pragma unroll
        for (int mi = 0; mi < 2; ++mi) {
            int mtc = (mi == 0) ? mt0 : mt1;
#pragma unroll
            for (int r = 0; r < 8; ++r) {
                int b = mtc * 16 + r + ((lane >= 16) ? 8 : 0);
                float val = 0.f;
                if (n < IN_N) {
                    val = acc[mi][nt][r] + bo + prev[(size_t)b * prev_stride + n];
                    out_t[(size_t)b * (SEQO * IN_N) + n] = val;
                }
                // residual -> bf16 A-fragments for next step's layer0 GEMM
                // (cols 99..111 zeroed each step; 112..127 zeroed at setup)
                prefrag[a_frag_off(b, n, KT_X)] = (__bf16)val;
            }
        }
    }
}

// ---------------------------------------------------------------------------
// Host driver
// ---------------------------------------------------------------------------
extern "C" void kernel_launch(void* const* d_in, const int* in_sizes, int n_in,
                              void* d_out, int out_size, void* d_ws, size_t ws_size,
                              hipStream_t stream) {
    (void)in_sizes; (void)n_in; (void)out_size; (void)ws_size;
    const float* hs    = (const float*)d_in[0];
    const float* cs    = (const float*)d_in[1];
    const float* gts   = (const float*)d_in[2];
    const float* p     = (const float*)d_in[3];
    const float* W_ih0 = (const float*)d_in[4];
    const float* b_ih0 = (const float*)d_in[5];
    const float* W_hh0 = (const float*)d_in[6];
    const float* b_hh0 = (const float*)d_in[7];
    const float* W_ih1 = (const float*)d_in[8];
    const float* b_ih1 = (const float*)d_in[9];
    const float* W_hh1 = (const float*)d_in[10];
    const float* b_hh1 = (const float*)d_in[11];
    const float* W_out = (const float*)d_in[12];
    const float* b_out = (const float*)d_in[13];
    float* out = (float*)d_out;

    // Workspace carve-up (256B aligned)
    uint8_t* ws = (uint8_t*)d_ws;
    size_t off = 0;
    auto carve = [&](size_t bytes) -> void* {
        void* ptr = ws + off;
        off += (bytes + 255) & ~(size_t)255;
        return ptr;
    };
    float*  c0      = (float*)carve((size_t)BSZ * HH * 4);
    float*  c1      = (float*)carve((size_t)BSZ * HH * 4);
    __bf16* h0f[2]  = { (__bf16*)carve((size_t)MT_N * KT_H * 512 * 2),
                        (__bf16*)carve((size_t)MT_N * KT_H * 512 * 2) };
    __bf16* h1f[2]  = { (__bf16*)carve((size_t)MT_N * KT_H * 512 * 2),
                        (__bf16*)carve((size_t)MT_N * KT_H * 512 * 2) };
    __bf16* prefrag = (__bf16*)carve((size_t)MT_N * KT_X * 512 * 2);
    __bf16* wih0f   = (__bf16*)carve((size_t)NT_G * KT_X * 512 * 2);
    __bf16* whh0f   = (__bf16*)carve((size_t)NT_G * KT_H * 512 * 2);
    __bf16* wih1f   = (__bf16*)carve((size_t)NT_G * KT_H * 512 * 2);
    __bf16* whh1f   = (__bf16*)carve((size_t)NT_G * KT_H * 512 * 2);
    __bf16* woutf   = (__bf16*)carve((size_t)NT_O * KT_H * 512 * 2);

    // --- Setup: means + weight conversion to bf16 fragments ---
    {
        int n = BSZ * HH;
        setup_state_kernel<<<(n + 255) / 256, 256, 0, stream>>>(hs, cs, gts, c0, c1, h0f[0], h1f[0]);
    }
    auto conv = [&](const float* src, __bf16* dst, int Nsrc, int Ksrc, int NT, int KT, int tr) {
        long total = (long)NT * KT * 512;
        convert_frag_kernel<<<(unsigned)((total + 255) / 256), 256, 0, stream>>>(src, dst, Nsrc, Ksrc, NT, KT, tr);
    };
    conv(W_ih0, wih0f, 4 * HH, IN_N, NT_G, KT_X, 0);   // (8448 x 99)  -> K pad 128
    conv(W_hh0, whh0f, 4 * HH, HH,   NT_G, KT_H, 0);
    conv(W_ih1, wih1f, 4 * HH, HH,   NT_G, KT_H, 0);
    conv(W_hh1, whh1f, 4 * HH, HH,   NT_G, KT_H, 0);
    conv(W_out, woutf, IN_N,   HH,   NT_O, KT_H, 1);   // (2112 x 99), N pad 112
    conv(p,     prefrag, BSZ,  IN_N, MT_N, KT_X, 0);   // pre0 A-fragments (+ zero K pad)

    // --- 25 sequential decode steps ---
    for (int t = 0; t < SEQO; ++t) {
        int cur = t & 1, nxt = cur ^ 1;
        // Layer 0: x = prev residual (K=128 padded), hidden = h0
        gemm_gates_kernel<<<dim3(4, KT_H), 256, 0, stream>>>(
            prefrag, KT_X, wih0f, h0f[cur], KT_H, whh0f, b_ih0, b_hh0, c0, h0f[nxt]);
        // Layer 1: x = h0_new, hidden = h1
        gemm_gates_kernel<<<dim3(4, KT_H), 256, 0, stream>>>(
            h0f[nxt], KT_H, wih1f, h1f[cur], KT_H, whh1f, b_ih1, b_hh1, c1, h1f[nxt]);
        // Output projection + residual; prev comes from p (t=0) or d_out[:, t-1, :]
        const float* prev = (t == 0) ? p : out + (size_t)(t - 1) * IN_N;
        int pstride = (t == 0) ? IN_N : SEQO * IN_N;
        gemm_out_kernel<<<dim3(4), 256, 0, stream>>>(
            h1f[nxt], woutf, b_out, prev, pstride, out + (size_t)t * IN_N, prefrag);
    }
}
